// SpatialPatchInteraction_83966610637179
// MI455X (gfx1250) — matrix operand order repaired and checked by reading, not compiled, not run on hardware
//
#include <hip/hip_runtime.h>
#include <math.h>

#define NTOK 4096
#define DMODEL 512
#define NHEAD 4
#define HD 128
#define NBINS 32
#define LUTN 4096

typedef __attribute__((ext_vector_type(16))) __bf16 v16bf;
typedef __attribute__((ext_vector_type(8)))  float  v8f;

union FragBF { v16bf v; uint4 q[2]; };

__device__ __forceinline__ unsigned short f2bf(float f) {
  unsigned int u = __float_as_uint(f);
  u += 0x7FFFu + ((u >> 16) & 1u);   // round-to-nearest-even
  return (unsigned short)(u >> 16);
}

// ---------------- f32 -> bf16 convert ----------------
__global__ void cvt_bf16_kernel(const float* __restrict__ src,
                                unsigned short* __restrict__ dst, int n) {
  int i = blockIdx.x * blockDim.x + threadIdx.x;
  if (i < n) dst[i] = f2bf(src[i]);
}

// ---------------- distance-bin LUT: bin(d^2) ----------------
__global__ void binlut_kernel(int* __restrict__ lut) {
  int i = blockIdx.x * blockDim.x + threadIdx.x;
  if (i >= LUTN) return;
  float dist = sqrtf((float)i);
  float mld = log1pf(40.0f * sqrtf(2.0f));
  int b = (int)(log1pf(dist) / mld * (float)(NBINS - 1));
  b = b < 0 ? 0 : (b > NBINS - 1 ? NBINS - 1 : b);
  lut[i] = b;
}

// ---------------- sinusoidal positional encoding (bf16) ----------------
__global__ void pe_kernel(const int* __restrict__ prow, const int* __restrict__ pcol,
                          unsigned short* __restrict__ pe) {
  int tid = blockIdx.x * blockDim.x + threadIdx.x;   // NTOK * 256 threads
  if (tid >= NTOK * 256) return;
  int n = tid >> 8, t = tid & 255;
  int useCols = t >> 7, tt = t & 127;
  float pos = (float)(useCols ? pcol[n] : prow[n]);
  float freq = expf((float)(2 * tt) * (-9.210340371976184f / 256.0f)); // -ln(1e4)/half
  float ang = pos * freq;
  int base = n * DMODEL + useCols * 256 + 2 * tt;
  pe[base]     = f2bf(sinf(ang));
  pe[base + 1] = f2bf(cosf(ang));
}

// ---------------- generic WMMA GEMM: C = scale*(A @ W^T) + bias ----------------
// A: M x K (bf16 rows), W: Nc x K (bf16 rows).  One wave per 16x16 C tile.
__global__ void gemm_wmma(const unsigned short* __restrict__ A,
                          const unsigned short* __restrict__ W,
                          const float* __restrict__ bias, float scale,
                          unsigned short* __restrict__ outB,
                          float* __restrict__ outF, int storeT,
                          int M, int Nc, int K) {
  int wid = threadIdx.x >> 5, lane = threadIdx.x & 31;
  int tile = blockIdx.x * (blockDim.x >> 5) + wid;
  int ntn = Nc >> 4;
  int tm = tile / ntn, tn = tile % ntn;
  if (tm * 16 >= M) return;                       // wave-uniform
  int half = lane >> 4, n15 = lane & 15;

  const unsigned short* Ap = A + (size_t)(tm * 16 + n15) * K + half * 8;
  const unsigned short* Wp = W + (size_t)(tn * 16 + n15) * K + half * 8;

  v8f c = {};
  for (int k0 = 0; k0 < K; k0 += 32) {
    FragBF a, b;
    a.q[0] = *(const uint4*)(Ap + k0);
    a.q[1] = *(const uint4*)(Ap + k0 + 16);
    b.q[0] = *(const uint4*)(Wp + k0);
    b.q[1] = *(const uint4*)(Wp + k0 + 16);
    c = __builtin_amdgcn_wmma_f32_16x16x32_bf16(false, a.v, false, b.v,
                                                (short)0, c, false, false);
  }
#pragma unroll
  for (int g = 0; g < 8; ++g) {
    int row = tm * 16 + g + 8 * half;
    int col = tn * 16 + n15;
    float v = c[g] * scale;
    if (bias) v += bias[(size_t)row * Nc + col];
    if (outF) outF[(size_t)row * Nc + col] = v;
    if (outB) {
      if (storeT) outB[(size_t)col * M + row] = f2bf(v);
      else        outB[(size_t)row * Nc + col] = f2bf(v);
    }
  }
}

// ---------------- attention pass 1: per-row max / sum-exp ----------------
__global__ void attn_stats(const unsigned short* __restrict__ Qb,
                           const unsigned short* __restrict__ Kb,
                           const int* __restrict__ prow, const int* __restrict__ pcol,
                           const float* __restrict__ dist_bias,
                           const int* __restrict__ lut,
                           float* __restrict__ Mbuf, float* __restrict__ Lbuf) {
  int wid = threadIdx.x >> 5, lane = threadIdx.x & 31;
  int tile = blockIdx.x * (blockDim.x >> 5) + wid;   // H * (N/16) waves
  int h = tile >> 8, qt = tile & 255;
  int i0 = qt * 16;
  int half = lane >> 4, n15 = lane & 15;

  const unsigned short* Qp = Qb + (size_t)(i0 + n15) * DMODEL + h * HD + half * 8;
  FragBF qa[4];
#pragma unroll
  for (int kc = 0; kc < 4; ++kc) {
    qa[kc].q[0] = *(const uint4*)(Qp + kc * 32);
    qa[kc].q[1] = *(const uint4*)(Qp + kc * 32 + 16);
  }
  int ri[8], ci[8];
#pragma unroll
  for (int g = 0; g < 8; ++g) { int r = i0 + g + 8 * half; ri[g] = prow[r]; ci[g] = pcol[r]; }

  float mrun[8], lrun[8];
#pragma unroll
  for (int g = 0; g < 8; ++g) { mrun[g] = -1e30f; lrun[g] = 0.0f; }

  const unsigned short* Kbase = Kb + h * HD + half * 8;
  for (int jt = 0; jt < NTOK / 16; ++jt) {
    int j0 = jt * 16;
    const unsigned short* Kp = Kbase + (size_t)(j0 + n15) * DMODEL;
    v8f c = {};
#pragma unroll
    for (int kc = 0; kc < 4; ++kc) {
      FragBF b;
      b.q[0] = *(const uint4*)(Kp + kc * 32);
      b.q[1] = *(const uint4*)(Kp + kc * 32 + 16);
      c = __builtin_amdgcn_wmma_f32_16x16x32_bf16(false, qa[kc].v, false, b.v,
                                                  (short)0, c, false, false);
    }
    int rj = prow[j0 + n15], cj = pcol[j0 + n15];
#pragma unroll
    for (int g = 0; g < 8; ++g) {
      int dr = ri[g] - rj, dc = ci[g] - cj;
      float s = c[g] + dist_bias[h * NBINS + lut[dr * dr + dc * dc]];
      float m = s;
      for (int off = 1; off < 16; off <<= 1) m = fmaxf(m, __shfl_xor(m, off));
      float p = expf(s - m);
      float sum = p;
      for (int off = 1; off < 16; off <<= 1) sum += __shfl_xor(sum, off);
      float mn = fmaxf(mrun[g], m);
      lrun[g] = lrun[g] * expf(mrun[g] - mn) + sum * expf(m - mn);
      mrun[g] = mn;
    }
  }
  if (n15 == 0) {
#pragma unroll
    for (int g = 0; g < 8; ++g) {
      int r = i0 + g + 8 * half;
      Mbuf[h * NTOK + r] = mrun[g];
      Lbuf[h * NTOK + r] = lrun[g];
    }
  }
}

// ---------------- attention pass 2: O = softmax(S) @ V ----------------
__global__ void attn_out(const unsigned short* __restrict__ Qb,
                         const unsigned short* __restrict__ Kb,
                         const unsigned short* __restrict__ Vtb,  // [DMODEL][NTOK]
                         const int* __restrict__ prow, const int* __restrict__ pcol,
                         const float* __restrict__ dist_bias,
                         const int* __restrict__ lut,
                         const float* __restrict__ Mbuf, const float* __restrict__ Lbuf,
                         unsigned short* __restrict__ AOb) {
  __shared__ unsigned short Pbuf[4][16][32];       // per-wave P staging (bf16 bits)
  int wid = threadIdx.x >> 5, lane = threadIdx.x & 31;
  int tile = blockIdx.x * (blockDim.x >> 5) + wid;
  int h = tile >> 8, qt = tile & 255;
  int i0 = qt * 16;
  int half = lane >> 4, n15 = lane & 15;

  const unsigned short* Qp = Qb + (size_t)(i0 + n15) * DMODEL + h * HD + half * 8;
  FragBF qa[4];
#pragma unroll
  for (int kc = 0; kc < 4; ++kc) {
    qa[kc].q[0] = *(const uint4*)(Qp + kc * 32);
    qa[kc].q[1] = *(const uint4*)(Qp + kc * 32 + 16);
  }
  int ri[8], ci[8];
  float Mrow[8];
#pragma unroll
  for (int g = 0; g < 8; ++g) {
    int r = i0 + g + 8 * half;
    ri[g] = prow[r]; ci[g] = pcol[r];
    Mrow[g] = Mbuf[h * NTOK + r];
  }
  v8f acc[8] = {};

  const unsigned short* Kbase = Kb + h * HD + half * 8;
  const unsigned short* Vbase = Vtb + (size_t)(h * HD + n15) * NTOK + half * 8;

  for (int jj = 0; jj < NTOK; jj += 32) {
#pragma unroll
    for (int sub = 0; sub < 2; ++sub) {
      int j0 = jj + sub * 16;
      const unsigned short* Kp = Kbase + (size_t)(j0 + n15) * DMODEL;
      v8f c = {};
#pragma unroll
      for (int kc = 0; kc < 4; ++kc) {
        FragBF b;
        b.q[0] = *(const uint4*)(Kp + kc * 32);
        b.q[1] = *(const uint4*)(Kp + kc * 32 + 16);
        c = __builtin_amdgcn_wmma_f32_16x16x32_bf16(false, qa[kc].v, false, b.v,
                                                    (short)0, c, false, false);
      }
      int rj = prow[j0 + n15], cj = pcol[j0 + n15];
#pragma unroll
      for (int g = 0; g < 8; ++g) {
        int dr = ri[g] - rj, dc = ci[g] - cj;
        float s = c[g] + dist_bias[h * NBINS + lut[dr * dr + dc * dc]];
        Pbuf[wid][g + 8 * half][sub * 16 + n15] = f2bf(expf(s - Mrow[g]));
      }
    }
    // all cross-lane P values must land in LDS before the fragment reload
    asm volatile("s_wait_dscnt 0" ::: "memory");

    FragBF pa;
    const unsigned short* Pp = &Pbuf[wid][n15][half * 8];
    pa.q[0] = *(const uint4*)(Pp);
    pa.q[1] = *(const uint4*)(Pp + 16);

#pragma unroll
    for (int dc = 0; dc < 8; ++dc) {
      FragBF b;
      const unsigned short* Vp = Vbase + (size_t)(dc * 16) * NTOK + jj;
      b.q[0] = *(const uint4*)(Vp);
      b.q[1] = *(const uint4*)(Vp + 16);
      acc[dc] = __builtin_amdgcn_wmma_f32_16x16x32_bf16(false, pa.v, false, b.v,
                                                        (short)0, acc[dc], false, false);
    }
  }

  float Lrow[8];
#pragma unroll
  for (int g = 0; g < 8; ++g) Lrow[g] = Lbuf[h * NTOK + i0 + g + 8 * half];
#pragma unroll
  for (int dc = 0; dc < 8; ++dc)
#pragma unroll
    for (int g = 0; g < 8; ++g) {
      int row = i0 + g + 8 * half;
      int col = h * HD + dc * 16 + n15;
      AOb[(size_t)row * DMODEL + col] = f2bf(acc[dc][g] / Lrow[g]);
    }
}

// ---------------- LayerNorm + softplus-gated residual ----------------
__global__ void ln_residual(const float* __restrict__ o, const float* __restrict__ vh,
                            const float* __restrict__ lnw, const float* __restrict__ lnb,
                            const float* __restrict__ gp, float* __restrict__ out) {
  __shared__ float sh1[256], sh2[256];
  int row = blockIdx.x, tid = threadIdx.x;
  float s = 0.f, s2 = 0.f;
  for (int j = tid; j < DMODEL; j += blockDim.x) {
    float v = o[(size_t)row * DMODEL + j];
    s += v; s2 += v * v;
  }
  sh1[tid] = s; sh2[tid] = s2; __syncthreads();
  for (int off = 128; off > 0; off >>= 1) {
    if (tid < off) { sh1[tid] += sh1[tid + off]; sh2[tid] += sh2[tid + off]; }
    __syncthreads();
  }
  float mu = sh1[0] / (float)DMODEL;
  float var = sh2[0] / (float)DMODEL - mu * mu;
  float inv = rsqrtf(var + 1e-5f);
  float gamma = log1pf(expf(gp[0]));
  for (int j = tid; j < DMODEL; j += blockDim.x) {
    float v = o[(size_t)row * DMODEL + j];
    out[(size_t)row * DMODEL + j] =
        vh[(size_t)row * DMODEL + j] + gamma * ((v - mu) * inv * lnw[j] + lnb[j]);
  }
}

extern "C" void kernel_launch(void* const* d_in, const int* in_sizes, int n_in,
                              void* d_out, int out_size, void* d_ws, size_t ws_size,
                              hipStream_t stream) {
  (void)in_sizes; (void)n_in; (void)out_size; (void)ws_size;
  const float* vision_h  = (const float*)d_in[0];
  const int*   prow      = (const int*)d_in[1];
  const int*   pcol      = (const int*)d_in[2];
  const float* pos_w     = (const float*)d_in[3];
  const float* q_w       = (const float*)d_in[4];
  const float* k_w       = (const float*)d_in[5];
  const float* v_w       = (const float*)d_in[6];
  const float* out_w     = (const float*)d_in[7];
  const float* dist_bias = (const float*)d_in[8];
  const float* gamma_p   = (const float*)d_in[9];
  const float* ln_w      = (const float*)d_in[10];
  const float* ln_b      = (const float*)d_in[11];
  float* out = (float*)d_out;

  char* p = (char*)d_ws;
  auto carve = [&](size_t bytes) -> char* {
    char* r = p; p += (bytes + 255) & ~(size_t)255; return r;
  };
  const size_t ND = (size_t)NTOK * DMODEL, DD = (size_t)DMODEL * DMODEL;
  unsigned short* vh_b   = (unsigned short*)carve(ND * 2);
  unsigned short* pe_b   = (unsigned short*)carve(ND * 2);
  unsigned short* hpos_b = (unsigned short*)carve(ND * 2);
  unsigned short* Qb     = (unsigned short*)carve(ND * 2);
  unsigned short* Kb     = (unsigned short*)carve(ND * 2);
  unsigned short* Vtb    = (unsigned short*)carve(ND * 2);
  unsigned short* AOb    = (unsigned short*)carve(ND * 2);
  unsigned short* posw_b = (unsigned short*)carve(DD * 2);
  unsigned short* qw_b   = (unsigned short*)carve(DD * 2);
  unsigned short* kw_b   = (unsigned short*)carve(DD * 2);
  unsigned short* vw_b   = (unsigned short*)carve(DD * 2);
  unsigned short* outw_b = (unsigned short*)carve(DD * 2);
  float* o_f   = (float*)carve(ND * 4);
  float* Mbuf  = (float*)carve((size_t)NHEAD * NTOK * 4);
  float* Lbuf  = (float*)carve((size_t)NHEAD * NTOK * 4);
  int*   lut   = (int*)carve((size_t)LUTN * 4);

  // 1) conversions + LUT + PE
  cvt_bf16_kernel<<<(int)((ND + 255) / 256), 256, 0, stream>>>(vision_h, vh_b, (int)ND);
  cvt_bf16_kernel<<<(int)((DD + 255) / 256), 256, 0, stream>>>(pos_w, posw_b, (int)DD);
  cvt_bf16_kernel<<<(int)((DD + 255) / 256), 256, 0, stream>>>(q_w, qw_b, (int)DD);
  cvt_bf16_kernel<<<(int)((DD + 255) / 256), 256, 0, stream>>>(k_w, kw_b, (int)DD);
  cvt_bf16_kernel<<<(int)((DD + 255) / 256), 256, 0, stream>>>(v_w, vw_b, (int)DD);
  cvt_bf16_kernel<<<(int)((DD + 255) / 256), 256, 0, stream>>>(out_w, outw_b, (int)DD);
  binlut_kernel<<<LUTN / 256, 256, 0, stream>>>(lut);
  pe_kernel<<<(NTOK * 256) / 256, 256, 0, stream>>>(prow, pcol, pe_b);

  // 2) projections (one wave per 16x16 tile; 4 waves / block)
  const int tiles = (NTOK / 16) * (DMODEL / 16);   // 8192
  const int gblk = tiles / 4;
  const float qscale = 0.08838834764831845f;       // 1/sqrt(128)
  gemm_wmma<<<gblk, 128, 0, stream>>>(pe_b, posw_b, vision_h, 1.0f,
                                      hpos_b, nullptr, 0, NTOK, DMODEL, DMODEL);
  gemm_wmma<<<gblk, 128, 0, stream>>>(hpos_b, qw_b, nullptr, qscale,
                                      Qb, nullptr, 0, NTOK, DMODEL, DMODEL);
  gemm_wmma<<<gblk, 128, 0, stream>>>(hpos_b, kw_b, nullptr, 1.0f,
                                      Kb, nullptr, 0, NTOK, DMODEL, DMODEL);
  gemm_wmma<<<gblk, 128, 0, stream>>>(vh_b, vw_b, nullptr, 1.0f,
                                      Vtb, nullptr, 1 /*transposed*/, NTOK, DMODEL, DMODEL);

  // 3) attention (H * N/16 = 1024 waves, 4 waves / block)
  attn_stats<<<256, 128, 0, stream>>>(Qb, Kb, prow, pcol, dist_bias, lut, Mbuf, Lbuf);
  attn_out<<<256, 128, 0, stream>>>(Qb, Kb, Vtb, prow, pcol, dist_bias, lut,
                                    Mbuf, Lbuf, AOb);

  // 4) output projection + LayerNorm residual
  gemm_wmma<<<gblk, 128, 0, stream>>>(AOb, outw_b, nullptr, 1.0f,
                                      nullptr, o_f, 0, NTOK, DMODEL, DMODEL);
  ln_residual<<<NTOK, 256, 0, stream>>>(o_f, vision_h, ln_w, ln_b, gamma_p, out);
}
